// PredRNNv2_model_82377472737849
// MI455X (gfx1250) — matrix operand order repaired and checked by reading, not compiled, not run on hardware
//
#include <hip/hip_runtime.h>
#include <hip/hip_bf16.h>

typedef _Float16 half_t;
typedef __attribute__((ext_vector_type(16))) _Float16 v16h;
typedef __attribute__((ext_vector_type(8)))  float    v8f;
typedef __attribute__((ext_vector_type(4))) unsigned int u32x4;
typedef __attribute__((ext_vector_type(8))) int          i32x8;
typedef __attribute__((ext_vector_type(4))) int          i32x4;

// ---------------- problem constants ----------------
constexpr int Bn  = 8;      // batch
constexpr int Tt  = 12;     // total frames
constexpr int WID = 32;     // H = W
constexpr int CHn = 16;     // frame channels
constexpr int NHn = 64;     // hidden channels per layer
constexpr int S   = WID * WID;       // 1024 spatial
constexpr int Mrows = Bn * S;        // 8192 = GEMM M
constexpr float FORGET_BIAS = 1.0f;
constexpr size_t FRAMES_OUT = (size_t)Bn * (Tt - 1) * S * CHn;   // 1441792

__device__ __forceinline__ float sigmoidf_(float x) { return 1.f / (1.f + __expf(-x)); }

// ---------------- tiny elementwise kernels ----------------
__global__ void fill0_kernel(float* p, int n) {
    int i = blockIdx.x * blockDim.x + threadIdx.x;
    if (i < n) p[i] = 0.f;
}

__global__ void f32_to_f16_kernel(const float* __restrict__ x, half_t* __restrict__ y, int n) {
    int i = blockIdx.x * blockDim.x + threadIdx.x;
    if (i < n) y[i] = (half_t)x[i];
}

// weights [Cout, Kr] (f32, flat OIHW) -> transposed f16 [Kpad][Cout], zero-padded K rows
__global__ void wcvt_kernel(const float* __restrict__ w, half_t* __restrict__ wt,
                            int Cout, int Kr, int Kpad) {
    int i = blockIdx.x * blockDim.x + threadIdx.x;
    int total = Kpad * Cout;
    if (i >= total) return;
    int k = i / Cout;
    int n = i - k * Cout;
    wt[i] = (k < Kr) ? (half_t)w[(size_t)n * Kr + k] : (half_t)0.f;
}

// ---------------- implicit-GEMM conv via WMMA + TDM weight staging ----------------
// x  : f16 [B, Cin, 32, 32]
// wt : f16 [Kpad, Cout]   (K = Cin*KSIZE*KSIZE padded to mult of 32, zero-filled)
// out: f32 [B, Cout, 32, 32]
// grid = (Mrows/128, ceil(Cout/64)), block = 256 (8 waves)
template <int KSIZE>
__global__ __launch_bounds__(256)
void conv_wmma_kernel(const half_t* __restrict__ xh, const half_t* __restrict__ wt,
                      float* __restrict__ out, int Cin, int Cout) {
    constexpr int KK   = KSIZE * KSIZE;
    constexpr int PADK = KSIZE / 2;
    const int Kr   = Cin * KK;
    const int Kpad = (Kr + 31) & ~31;

    // single shared block so the Bs byte offset is deterministic for the TDM descriptor
    __shared__ half_t smem[128 * 40 + 32 * 72];
    half_t* As = smem;                 // A tile [128 rows][32 k], row stride 40 halves
    half_t* Bs = smem + 128 * 40;      // B tile [32 k][64 n],  row stride 72 halves
    constexpr unsigned BS_LDS_OFF = 128u * 40u * 2u;   // byte offset of Bs

    const int tid  = threadIdx.x;
    const int wave = tid >> 5;
    const int lane = tid & 31;
    const int m0 = blockIdx.x * 128;
    const int n0 = blockIdx.y * 64;

    // A-staging assignment: thread -> (row, half of the 32-wide k slab)
    const int a_mi  = tid >> 1;            // 0..127
    const int a_kb  = (tid & 1) << 4;      // 0 or 16
    const int a_m   = m0 + a_mi;
    const int a_b   = a_m >> 10;
    const int a_pos = a_m & 1023;
    const int a_y   = a_pos >> 5;
    const int a_x   = a_pos & 31;
    const half_t* xb = xh + ((size_t)(a_b * Cin) << 10);

    v8f acc[4] = {};

    // per-lane fragment addressing (ISA 16-bit A layout: lanes<16 hold K {0..7,16..23})
    const int mr = (wave << 4) + (lane & 15);
    const int kg = (lane < 16) ? 0 : 8;

    for (int kk = 0; kk < Kpad; kk += 32) {
        // ---- kick off weight tile DMA: global [32 x Cout-clamped-64] -> LDS Bs ----
        if (wave == 0) {
            unsigned long long gaddr =
                (unsigned long long)(const void*)(wt + (size_t)kk * Cout + n0);
            u32x4 g0;
            g0[0] = 1u;                                     // count = 1 valid descriptor
            g0[1] = BS_LDS_OFF;                             // lds_addr (bytes)
            g0[2] = (unsigned)(gaddr & 0xffffffffull);      // global_addr[31:0]
            g0[3] = (unsigned)((gaddr >> 32) & 0x1ffffffull) | (2u << 30); // [56:32]|type=2
            i32x8 g1;
            // data_size=2B (code 1), pad_enable, pad_interval=32 DW (code 4),
            // pad_amount=4 DW (code 3) -> LDS row stride 128B + 16B = 72 halves
            g1[0] = (int)((1u << 16) | (1u << 20) | (4u << 22) | (3u << 25));
            g1[1] = (int)(((unsigned)Cout & 0xffffu) << 16);                  // tensor_dim0 lo16
            g1[2] = (int)((((unsigned)Cout >> 16) & 0xffffu) | (32u << 16));  // dim0 hi, tensor_dim1=32
            g1[3] = (int)(64u << 16);                       // dim1 hi=0, tile_dim0=64
            g1[4] = 32;                                     // tile_dim1=32, tile_dim2=0
            g1[5] = Cout;                                   // tensor_dim0_stride lo32 (elements)
            g1[6] = 0;
            g1[7] = 0;
            i32x4 z4 = {0, 0, 0, 0};
#if __clang_major__ >= 23
            i32x8 z8 = {0, 0, 0, 0, 0, 0, 0, 0};
            __builtin_amdgcn_tensor_load_to_lds(g0, g1, z4, z4, z8, 0);
#else
            __builtin_amdgcn_tensor_load_to_lds(g0, g1, z4, z4, 0);
#endif
        }

        // ---- stage A tile: f16 im2col gather (overlaps with the TDM transfer) ----
        if (KSIZE > 1)
            __builtin_prefetch(xb + ((((kk + 32) / KK) << 10) + a_pos), 0, 0); // next channel slab
        #pragma unroll
        for (int j = 0; j < 16; ++j) {
            int k = kk + a_kb + j;
            half_t v = (half_t)0.f;
            if (k < Kr) {
                int ci, ky, kx;
                if (KSIZE == 1) { ci = k; ky = 0; kx = 0; }
                else { ci = k / KK; int r = k - ci * KK; ky = r / KSIZE; kx = r - ky * KSIZE; }
                int yy = a_y + ky - PADK;
                int xx = a_x + kx - PADK;
                if ((unsigned)yy < 32u && (unsigned)xx < 32u)
                    v = xb[(ci << 10) + (yy << 5) + xx];
            }
            As[a_mi * 40 + a_kb + j] = v;
        }

        if (wave == 0) __builtin_amdgcn_s_wait_tensorcnt(0);   // weight tile landed in LDS
        __syncthreads();

        // ---- load fragments, 4 WMMAs per wave per K-step ----
        v16h afrag;
        {
            const half_t* ap = &As[mr * 40];
            #pragma unroll
            for (int i = 0; i < 8; ++i) { afrag[i] = ap[kg + i]; afrag[8 + i] = ap[16 + kg + i]; }
        }
        #pragma unroll
        for (int s = 0; s < 4; ++s) {
            const half_t* bp = &Bs[(s << 4) + (lane & 15)];
            v16h bfrag;
            #pragma unroll
            for (int i = 0; i < 8; ++i) { bfrag[i] = bp[(kg + i) * 72]; bfrag[8 + i] = bp[(16 + kg + i) * 72]; }
            acc[s] = __builtin_amdgcn_wmma_f32_16x16x32_f16(
                false, afrag, false, bfrag, (short)0, acc[s], false, false);
        }
        __syncthreads();   // everyone done with As/Bs before next DMA / restage
    }

    // ---- epilogue: D layout lanes<16 -> M=v, lanes>=16 -> M=8+v; N = lane&15 ----
    const int n_l  = lane & 15;
    const int m_hi = (lane < 16) ? 0 : 8;
    #pragma unroll
    for (int s = 0; s < 4; ++s) {
        int n = n0 + (s << 4) + n_l;
        if (n < Cout) {
            #pragma unroll
            for (int v = 0; v < 8; ++v) {
                int m   = m0 + (wave << 4) + m_hi + v;
                int b   = m >> 10;
                int pos = m & 1023;
                out[((size_t)(b * Cout + n) << 10) + pos] = acc[s][v];
            }
        }
    }
}

// ---------------- LayerNorm over (C,H,W) per sample, in place ----------------
__global__ __launch_bounds__(256)
void layernorm_kernel(float* __restrict__ x, const float* __restrict__ g,
                      const float* __restrict__ b, int C) {
    const int bi = blockIdx.x;
    const int n = C << 10;
    float* xp = x + (size_t)bi * n;
    float s = 0.f, s2 = 0.f;
    for (int i = threadIdx.x; i < n; i += 256) { float v = xp[i]; s += v; s2 += v * v; }
    __shared__ float r0[256], r1[256];
    r0[threadIdx.x] = s; r1[threadIdx.x] = s2;
    __syncthreads();
    for (int off = 128; off > 0; off >>= 1) {
        if (threadIdx.x < off) {
            r0[threadIdx.x] += r0[threadIdx.x + off];
            r1[threadIdx.x] += r1[threadIdx.x + off];
        }
        __syncthreads();
    }
    float mean = r0[0] / n;
    float var  = r1[0] / n - mean * mean;
    float inv  = rsqrtf(var + 1e-5f);
    for (int i = threadIdx.x; i < n; i += 256)
        xp[i] = (xp[i] - mean) * inv * g[i] + b[i];
}

// ---------------- gate math (c/m update), writes dc/dm and mem=[c|m] ----------------
__global__ void gates_kernel(const float* __restrict__ xc, const float* __restrict__ hc,
                             const float* __restrict__ mc, float* __restrict__ c,
                             float* __restrict__ m, float* __restrict__ dc,
                             float* __restrict__ dm, float* __restrict__ mem) {
    const int CS = NHn << 10;                    // 65536
    int idx = blockIdx.x * blockDim.x + threadIdx.x;
    if (idx >= Bn * CS) return;
    int b = idx >> 16;
    int r = idx & (CS - 1);
    const float* xb = xc + (size_t)b * 7 * CS;
    const float* hb = hc + (size_t)b * 4 * CS;
    const float* mb = mc + (size_t)b * 3 * CS;
    float it = sigmoidf_(xb[r] + hb[r]);
    float ft = sigmoidf_(xb[CS + r] + hb[CS + r] + FORGET_BIAS);
    float gt = tanhf(xb[2 * CS + r] + hb[2 * CS + r]);
    float d_c = it * gt;
    float cn  = ft * c[idx] + d_c;
    float itp = sigmoidf_(xb[3 * CS + r] + mb[r]);
    float ftp = sigmoidf_(xb[4 * CS + r] + mb[CS + r] + FORGET_BIAS);
    float gtp = tanhf(xb[5 * CS + r] + mb[2 * CS + r]);
    float d_m = itp * gtp;
    float mn  = ftp * m[idx] + d_m;
    c[idx] = cn; m[idx] = mn; dc[idx] = d_c; dm[idx] = d_m;
    float* memb = mem + (size_t)b * 2 * CS;
    memb[r] = cn; memb[CS + r] = mn;
}

// ---------------- output gate: h = sigmoid(o_x+o_h+LN(conv(mem,Wo))) * tanh(conv1x1) ----
__global__ void outgate_kernel(const float* __restrict__ xc, const float* __restrict__ hc,
                               const float* __restrict__ oc, const float* __restrict__ cl,
                               float* __restrict__ h) {
    const int CS = NHn << 10;
    int idx = blockIdx.x * blockDim.x + threadIdx.x;
    if (idx >= Bn * CS) return;
    int b = idx >> 16;
    int r = idx & (CS - 1);
    float ox = xc[(size_t)b * 7 * CS + 6 * CS + r];
    float oh = hc[(size_t)b * 4 * CS + 3 * CS + r];
    float ot = sigmoidf_(ox + oh + oc[idx]);
    h[idx] = ot * tanhf(cl[idx]);
}

// ---------------- frame prep: net = (t==0)? frame : mask*frame + (1-mask)*xgen, -> f16 NCHW
__global__ void prep_kernel(const float* __restrict__ frames, const float* __restrict__ mask,
                            const float* __restrict__ xgen, half_t* __restrict__ net_h, int t) {
    const int CS = CHn << 10;
    int idx = blockIdx.x * blockDim.x + threadIdx.x;
    if (idx >= Bn * CS) return;
    int b = idx / CS;
    int r = idx - b * CS;
    int ch = r >> 10;
    int pos = r & 1023;
    float fr = frames[(((size_t)(b * Tt + t) << 10) + pos) * CHn + ch];
    float v;
    if (t == 0) v = fr;
    else {
        float mk = mask[(((size_t)(b * (Tt - 1) + (t - 1)) << 10) + pos) * CHn + ch];
        v = mk * fr + (1.f - mk) * xgen[idx];
    }
    net_h[idx] = (half_t)v;
}

// ---------------- write generated frame to output [B,T-1,H,W,C] ----------------
__global__ void write_frames_kernel(const float* __restrict__ xgen, float* __restrict__ out, int t) {
    const int CS = CHn << 10;
    int idx = blockIdx.x * blockDim.x + threadIdx.x;
    if (idx >= Bn * CS) return;
    int b = idx / CS;
    int r = idx - b * CS;
    int ch = r >> 10;
    int pos = r & 1023;
    out[(((size_t)(b * (Tt - 1) + t) << 10) + pos) * CHn + ch] = xgen[idx];
}

// ---------------- decouple loss: mean |cos(ac_row, am_row)| over (b,c) ----------------
__global__ __launch_bounds__(256)
void dls_kernel(const float* __restrict__ ac, const float* __restrict__ am,
                float* __restrict__ outslot) {
    int bc = blockIdx.x;                          // 0 .. B*NH-1 (=512)
    const float* a = ac + ((size_t)bc << 10);
    const float* c = am + ((size_t)bc << 10);
    float dot = 0.f, na = 0.f, nb = 0.f;
    for (int i = threadIdx.x; i < 1024; i += 256) {
        float x = a[i], y = c[i];
        dot += x * y; na += x * x; nb += y * y;
    }
    __shared__ float r0[256], r1[256], r2[256];
    r0[threadIdx.x] = dot; r1[threadIdx.x] = na; r2[threadIdx.x] = nb;
    __syncthreads();
    for (int off = 128; off > 0; off >>= 1) {
        if (threadIdx.x < off) {
            r0[threadIdx.x] += r0[threadIdx.x + off];
            r1[threadIdx.x] += r1[threadIdx.x + off];
            r2[threadIdx.x] += r2[threadIdx.x + off];
        }
        __syncthreads();
    }
    if (threadIdx.x == 0) {
        float d = r0[0] / (fmaxf(sqrtf(r1[0]), 1e-12f) * fmaxf(sqrtf(r2[0]), 1e-12f));
        atomicAdd(outslot, fabsf(d) * (1.f / (Bn * NHn)));
    }
}

// =====================================================================
extern "C" void kernel_launch(void* const* d_in, const int* in_sizes, int n_in,
                              void* d_out, int out_size, void* d_ws, size_t ws_size,
                              hipStream_t stream) {
    (void)in_sizes; (void)n_in; (void)out_size; (void)ws_size;

    const float* frames = (const float*)d_in[0];
    const float* mask   = (const float*)d_in[1];
    struct Cell { const float *Wx,*gx,*bx,*Wh,*gh,*bh,*Wm,*gm,*bm,*Wo,*go,*bo,*Wcl; };
    Cell cells[4];
    int p = 2;
    for (int i = 0; i < 4; ++i) {
        cells[i].Wx  = (const float*)d_in[p++]; cells[i].gx = (const float*)d_in[p++];
        cells[i].bx  = (const float*)d_in[p++]; cells[i].Wh = (const float*)d_in[p++];
        cells[i].gh  = (const float*)d_in[p++]; cells[i].bh = (const float*)d_in[p++];
        cells[i].Wm  = (const float*)d_in[p++]; cells[i].gm = (const float*)d_in[p++];
        cells[i].bm  = (const float*)d_in[p++]; cells[i].Wo = (const float*)d_in[p++];
        cells[i].go  = (const float*)d_in[p++]; cells[i].bo = (const float*)d_in[p++];
        cells[i].Wcl = (const float*)d_in[p++];
    }
    const float* conv_last_w = (const float*)d_in[p++];
    const float* adapter_w   = (const float*)d_in[p++];

    // -------- workspace carving --------
    char* wsp = (char*)d_ws;
    auto alloc_f = [&](size_t n) -> float* {
        float* r = (float*)wsp; wsp += ((n * 4 + 255) & ~(size_t)255); return r;
    };
    auto alloc_h = [&](size_t n) -> half_t* {
        half_t* r = (half_t*)wsp; wsp += ((n * 2 + 255) & ~(size_t)255); return r;
    };
    const size_t CS = (size_t)NHn * 1024;          // 65536 per sample per 64ch field
    float *h[4], *c[4];
    for (int i = 0; i < 4; ++i) { h[i] = alloc_f(Bn * CS); c[i] = alloc_f(Bn * CS); }
    float* memory = alloc_f(Bn * CS);
    float* xgen   = alloc_f((size_t)Bn * CHn * 1024);
    float* xc  = alloc_f(Bn * 7 * CS);
    float* hcb = alloc_f(Bn * 4 * CS);
    float* mcb = alloc_f(Bn * 3 * CS);
    float* memb = alloc_f(Bn * 2 * CS);
    float* oc = alloc_f(Bn * CS);
    float* cl = alloc_f(Bn * CS);
    float* dc = alloc_f(Bn * CS);
    float* dm = alloc_f(Bn * CS);
    float* ac = alloc_f(Bn * CS);
    float* am = alloc_f(Bn * CS);
    half_t* act_h = alloc_h(Bn * 2 * CS);          // max Cin = 128

    half_t *WxT[4], *WhT[4], *WmT[4], *WoT[4], *WclT[4];
    int KpX[4];
    for (int i = 0; i < 4; ++i) {
        int cin = (i == 0) ? CHn : NHn;
        int KrX = cin * 25; KpX[i] = (KrX + 31) & ~31;
        WxT[i]  = alloc_h((size_t)KpX[i] * (7 * NHn));
        WhT[i]  = alloc_h((size_t)1600 * (4 * NHn));
        WmT[i]  = alloc_h((size_t)1600 * (3 * NHn));
        WoT[i]  = alloc_h((size_t)3200 * NHn);
        WclT[i] = alloc_h((size_t)128 * NHn);
    }
    half_t* clastT = alloc_h((size_t)64 * CHn);
    half_t* adptT  = alloc_h((size_t)64 * NHn);

    auto g1 = [](size_t n) { return dim3((unsigned)((n + 255) / 256)); };

    // -------- convert weights to f16 [Kpad][Cout] once per launch --------
    for (int i = 0; i < 4; ++i) {
        int cin = (i == 0) ? CHn : NHn;
        int KrX = cin * 25;
        wcvt_kernel<<<g1((size_t)KpX[i] * 7 * NHn), 256, 0, stream>>>(cells[i].Wx, WxT[i], 7 * NHn, KrX, KpX[i]);
        wcvt_kernel<<<g1((size_t)1600 * 4 * NHn), 256, 0, stream>>>(cells[i].Wh, WhT[i], 4 * NHn, 1600, 1600);
        wcvt_kernel<<<g1((size_t)1600 * 3 * NHn), 256, 0, stream>>>(cells[i].Wm, WmT[i], 3 * NHn, 1600, 1600);
        wcvt_kernel<<<g1((size_t)3200 * NHn), 256, 0, stream>>>(cells[i].Wo, WoT[i], NHn, 3200, 3200);
        wcvt_kernel<<<g1((size_t)128 * NHn), 256, 0, stream>>>(cells[i].Wcl, WclT[i], NHn, 128, 128);
    }
    wcvt_kernel<<<g1((size_t)64 * CHn), 256, 0, stream>>>(conv_last_w, clastT, CHn, 64, 64);
    wcvt_kernel<<<g1((size_t)64 * NHn), 256, 0, stream>>>(adapter_w, adptT, NHn, 64, 64);

    // -------- zero-init recurrent state + dls accumulators --------
    for (int i = 0; i < 4; ++i) {
        fill0_kernel<<<g1(Bn * CS), 256, 0, stream>>>(h[i], (int)(Bn * CS));
        fill0_kernel<<<g1(Bn * CS), 256, 0, stream>>>(c[i], (int)(Bn * CS));
    }
    fill0_kernel<<<g1(Bn * CS), 256, 0, stream>>>(memory, (int)(Bn * CS));
    fill0_kernel<<<g1((size_t)Bn * CHn * 1024), 256, 0, stream>>>(xgen, Bn * CHn * 1024);
    fill0_kernel<<<dim3(1), 64, 0, stream>>>((float*)d_out + FRAMES_OUT, (Tt - 1) * 4);

    const dim3 cblk(256);
    // -------- time loop --------
    for (int t = 0; t < Tt - 1; ++t) {
        prep_kernel<<<g1((size_t)Bn * CHn * 1024), 256, 0, stream>>>(frames, mask, xgen, act_h, t);
        for (int i = 0; i < 4; ++i) {
            int cin = (i == 0) ? CHn : NHn;
            if (i > 0)
                f32_to_f16_kernel<<<g1((size_t)Bn * cin * 1024), 256, 0, stream>>>(h[i - 1], act_h, Bn * cin * 1024);
            // xc = LN(conv5(inp, Wx))
            conv_wmma_kernel<5><<<dim3(Mrows / 128, 7), cblk, 0, stream>>>(act_h, WxT[i], xc, cin, 7 * NHn);
            layernorm_kernel<<<dim3(Bn), cblk, 0, stream>>>(xc, cells[i].gx, cells[i].bx, 7 * NHn);
            // hc = LN(conv5(h, Wh))
            f32_to_f16_kernel<<<g1(Bn * CS), 256, 0, stream>>>(h[i], act_h, (int)(Bn * CS));
            conv_wmma_kernel<5><<<dim3(Mrows / 128, 4), cblk, 0, stream>>>(act_h, WhT[i], hcb, NHn, 4 * NHn);
            layernorm_kernel<<<dim3(Bn), cblk, 0, stream>>>(hcb, cells[i].gh, cells[i].bh, 4 * NHn);
            // mc = LN(conv5(m, Wm))
            f32_to_f16_kernel<<<g1(Bn * CS), 256, 0, stream>>>(memory, act_h, (int)(Bn * CS));
            conv_wmma_kernel<5><<<dim3(Mrows / 128, 3), cblk, 0, stream>>>(act_h, WmT[i], mcb, NHn, 3 * NHn);
            layernorm_kernel<<<dim3(Bn), cblk, 0, stream>>>(mcb, cells[i].gm, cells[i].bm, 3 * NHn);
            // c/m update
            gates_kernel<<<g1(Bn * CS), 256, 0, stream>>>(xc, hcb, mcb, c[i], memory, dc, dm, memb);
            // o-path conv + 1x1 conv on mem
            f32_to_f16_kernel<<<g1(Bn * 2 * CS), 256, 0, stream>>>(memb, act_h, (int)(Bn * 2 * CS));
            conv_wmma_kernel<5><<<dim3(Mrows / 128, 1), cblk, 0, stream>>>(act_h, WoT[i], oc, 2 * NHn, NHn);
            layernorm_kernel<<<dim3(Bn), cblk, 0, stream>>>(oc, cells[i].go, cells[i].bo, NHn);
            conv_wmma_kernel<1><<<dim3(Mrows / 128, 1), cblk, 0, stream>>>(act_h, WclT[i], cl, 2 * NHn, NHn);
            outgate_kernel<<<g1(Bn * CS), 256, 0, stream>>>(xc, hcb, oc, cl, h[i]);
            // decouple loss: adapter(dc) vs adapter(dm)
            f32_to_f16_kernel<<<g1(Bn * CS), 256, 0, stream>>>(dc, act_h, (int)(Bn * CS));
            conv_wmma_kernel<1><<<dim3(Mrows / 128, 1), cblk, 0, stream>>>(act_h, adptT, ac, NHn, NHn);
            f32_to_f16_kernel<<<g1(Bn * CS), 256, 0, stream>>>(dm, act_h, (int)(Bn * CS));
            conv_wmma_kernel<1><<<dim3(Mrows / 128, 1), cblk, 0, stream>>>(act_h, adptT, am, NHn, NHn);
            dls_kernel<<<dim3(Bn * NHn), cblk, 0, stream>>>(ac, am, (float*)d_out + FRAMES_OUT + (t * 4 + i));
        }
        // x_gen = conv1x1(h[3], conv_last)
        f32_to_f16_kernel<<<g1(Bn * CS), 256, 0, stream>>>(h[3], act_h, (int)(Bn * CS));
        conv_wmma_kernel<1><<<dim3(Mrows / 128, 1), cblk, 0, stream>>>(act_h, clastT, xgen, NHn, CHn);
        write_frames_kernel<<<g1((size_t)Bn * CHn * 1024), 256, 0, stream>>>(xgen, (float*)d_out, t);
    }
}
